// DCMCLITA_62431644615165
// MI455X (gfx1250) — matrix operand order repaired and checked by reading, not compiled, not run on hardware
//
#include <hip/hip_runtime.h>
#include <hip/hip_bf16.h>
#include <cstdint>
#include <math.h>

// ---------------------------------------------------------------------------
// Types for CDNA5 WMMA (wave32): 16x16x32 bf16 -> f32, and TDM descriptors
// ---------------------------------------------------------------------------
typedef __attribute__((ext_vector_type(16))) __bf16 v16bf;
typedef __attribute__((ext_vector_type(8)))  float  v8f;
typedef __attribute__((ext_vector_type(4)))  unsigned int v4ui;
typedef __attribute__((ext_vector_type(8)))  int v8si;
typedef __attribute__((ext_vector_type(4)))  int v4si;

union FragU {
    v16bf v;
    uint4 q[2];
};

__device__ __forceinline__ uint16_t f2bf(float x) {
    uint32_t u = __builtin_bit_cast(uint32_t, x);
    u += 0x7FFFu + ((u >> 16) & 1u);   // round-to-nearest-even
    return (uint16_t)(u >> 16);
}
__device__ __forceinline__ float bf2f(uint16_t h) {
    uint32_t u = ((uint32_t)h) << 16;
    return __builtin_bit_cast(float, u);
}
__device__ __forceinline__ float sigf(float x) { return 1.0f / (1.0f + expf(-x)); }

// A-matrix fragment (16x32 bf16, row-major source, stride lda elements).
__device__ __forceinline__ v16bf load_a_frag(const uint16_t* A, size_t lda,
                                             int rowbase, int kt) {
    int lane = threadIdx.x & 31;
    int row  = rowbase + (lane & 15);
    int kb   = (lane < 16) ? 0 : 8;
    const uint16_t* p = A + (size_t)row * lda + (size_t)kt * 32 + kb;
    FragU f;
    f.q[0] = *(const uint4*)(p);        // K = kb .. kb+7
    f.q[1] = *(const uint4*)(p + 16);   // K = kb+16 .. kb+23
    return f.v;
}

// B-matrix fragment from pre-swizzled pack: tile = 32 lanes x 16 bf16 contiguous.
__device__ __forceinline__ v16bf load_b_frag(const uint16_t* tile) {
    int lane = threadIdx.x & 31;
    const uint16_t* p = tile + lane * 16;
    FragU f;
    f.q[0] = *(const uint4*)(p);
    f.q[1] = *(const uint4*)(p + 8);
    return f.v;
}

// ---------------------------------------------------------------------------
// Tensor Data Mover: 2-D tile (rows of tile0_u8 x 8 bytes, tile1 rows with
// stride stride0_u8 x 8 bytes) from global -> contiguous LDS.
// D# layout per CDNA5 ISA (group0 128b, group1 256b).
// This toolchain exposes the 6-arg builtin:
//   (uint32x4 g0, int32x8 g1, int32x4 g2, int32x4 g3, int32x8 spare, i32 cpol)
// ---------------------------------------------------------------------------
#if __has_builtin(__builtin_amdgcn_tensor_load_to_lds) && __has_builtin(__builtin_amdgcn_s_wait_tensorcnt)
#define HAVE_TDM 1
#else
#define HAVE_TDM 0
#endif

#if HAVE_TDM
__device__ __forceinline__ void tdm_load_2d(const void* gsrc, void* ldst,
                                            unsigned tile0_u8, unsigned tile1,
                                            unsigned stride0_u8) {
    unsigned lds_off = (unsigned)(size_t)ldst;            // LDS byte offset (addr[31:0])
    unsigned long long ga = (unsigned long long)(size_t)gsrc;
    unsigned dim0 = stride0_u8 ? stride0_u8 : tile0_u8;   // tensor extent >= tile
    unsigned dim1 = (tile1 > 1) ? tile1 : 1u;
    v4ui g0 = { 1u,                                        // count=1, user descriptor
                lds_off,                                   // lds_addr
                (unsigned)ga,                              // global_addr[31:0]
                (unsigned)((ga >> 32) & 0x1FFFFFFu) | (2u << 30) }; // addr[56:32] | type=2
    v8si g1 = { (int)(3u << 16),                           // data_size = 8 bytes
                (int)((dim0 & 0xFFFFu) << 16),             // tensor_dim0[15:0]
                (int)((dim0 >> 16) | ((dim1 & 0xFFFFu) << 16)),   // dim0[31:16] | dim1[15:0]
                (int)((dim1 >> 16) | ((tile0_u8 & 0xFFFFu) << 16)), // dim1[31:16] | tile_dim0
                (int)((tile1 > 1) ? tile1 : 0u),           // tile_dim1 (0 = unused)
                (int)stride0_u8,                           // tensor_dim0_stride[31:0]
                0, 0 };
    v4si z4 = {0, 0, 0, 0};                                // groups 2/3: <=2-D tensor
    v8si z8 = {0, 0, 0, 0, 0, 0, 0, 0};
    __builtin_amdgcn_tensor_load_to_lds(g0, g1, z4, z4, z8, 0);
}
#endif

// ---------------------------------------------------------------------------
// Constants
// ---------------------------------------------------------------------------
#define BB 128
#define LL 512
#define CH 6
#define HH 256
#define GG 1024
#define FD 224          // 198 features padded to 224 (7 k-tiles of 32)
#define KT_IH 7
#define KT_HH 8
#define R_MERC 6378137.0f
#define R_EARTH 6371.0f
#define DEG2RAD 0.017453292519943295f

// ---------------------------------------------------------------------------
// Kernel: zero init
// ---------------------------------------------------------------------------
__global__ void k_zero_u32(unsigned* p, size_t n) {
    size_t i = (size_t)blockIdx.x * blockDim.x + threadIdx.x;
    size_t stride = (size_t)gridDim.x * blockDim.x;
    for (; i < n; i += stride) p[i] = 0u;
}

// ---------------------------------------------------------------------------
// Kernel: per-batch mercator normalize + speed/bearing fixup + x3 build
// ---------------------------------------------------------------------------
__device__ __forceinline__ float merc_x(float lon) { return R_MERC * (lon * DEG2RAD); }
__device__ __forceinline__ float merc_y(float lat) {
    return R_MERC * logf(tanf(0.78539816339f + (lat * DEG2RAD) * 0.5f));
}

__global__ __launch_bounds__(256)
void k_prep(const float* __restrict__ x1, const float* __restrict__ x2,
            const float* __restrict__ dtime,
            float* __restrict__ xn1, float* __restrict__ xn2,
            float* __restrict__ xn3) {
    __shared__ float sMxA[256], sMnA[256], sMxB[256], sMnB[256];
    __shared__ float sh[8];
    int b = blockIdx.x;
    int tid = threadIdx.x;

    float mxa = -3.4e38f, mna = 3.4e38f, mxb = -3.4e38f, mnb = 3.4e38f;
    for (int l = tid; l < LL; l += 256) {
        for (int s = 0; s < 2; ++s) {
            const float* xp = (s == 0 ? x1 : x2) + ((size_t)b * LL + l) * CH;
            float a  = merc_x(xp[1]);
            float bv = merc_y(xp[0]);
            mxa = fmaxf(mxa, a);  mna = fminf(mna, a);
            mxb = fmaxf(mxb, bv); mnb = fminf(mnb, bv);
        }
    }
    sMxA[tid] = mxa; sMnA[tid] = mna; sMxB[tid] = mxb; sMnB[tid] = mnb;
    __syncthreads();
    for (int st = 128; st > 0; st >>= 1) {
        if (tid < st) {
            sMxA[tid] = fmaxf(sMxA[tid], sMxA[tid + st]);
            sMnA[tid] = fminf(sMnA[tid], sMnA[tid + st]);
            sMxB[tid] = fmaxf(sMxB[tid], sMxB[tid + st]);
            sMnB[tid] = fminf(sMnB[tid], sMnB[tid + st]);
        }
        __syncthreads();
    }
    if (tid == 0) {
        float da = sMxA[0] - sMnA[0] + 1e-8f;
        float db = sMxB[0] - sMnB[0] + 1e-8f;
        sh[0] = sMnA[0]; sh[1] = sMnB[0];
        sh[2] = 1.0f / da; sh[3] = 1.0f / db;
        const float* p1 = x1 + ((size_t)b * LL + (LL - 1)) * CH;
        const float* p2 = x2 + ((size_t)b * LL + 0) * CH;
        float la1 = p1[0] * DEG2RAD, lo1 = p1[1] * DEG2RAD;
        float la2 = p2[0] * DEG2RAD, lo2 = p2[1] * DEG2RAD;
        float dlon = lo2 - lo1, dlat = la2 - la1;
        float sh1 = sinf(dlat * 0.5f), sh2 = sinf(dlon * 0.5f);
        float a = sh1 * sh1 + cosf(la1) * cosf(la2) * sh2 * sh2;
        float dist = 2.0f * asinf(sqrtf(fminf(fmaxf(a, 0.f), 1.f))) * R_EARTH;
        float yb = sinf(dlon) * cosf(la2);
        float xb = cosf(la1) * sinf(la2) - sinf(la1) * cosf(la2) * cosf(dlon);
        float deg = atan2f(yb, xb) * 57.29577951308232f + 360.0f;
        float r = fmodf(deg, 360.0f); if (r < 0.f) r += 360.0f;
        float brg = r * DEG2RAD;
        float dt = dtime[b]; if (dt == 0.f) dt = 1.0f;
        float spd = dist / dt * 1000.0f / 0.514444f;
        sh[4] = spd; sh[5] = spd * sinf(brg); sh[6] = spd * cosf(brg);
    }
    __syncthreads();
    float mnaF = sh[0], mnbF = sh[1], ida = sh[2], idb = sh[3];
    float spd = sh[4], vx = sh[5], vy = sh[6];

    for (int l = tid; l < LL; l += 256) {
        for (int s = 0; s < 2; ++s) {
            const float* xp = (s == 0 ? x1 : x2) + ((size_t)b * LL + l) * CH;
            float* op = (s == 0 ? xn1 : xn2) + ((size_t)b * LL + l) * CH;
            op[0] = (merc_x(xp[1]) - mnaF) * ida;
            op[1] = (merc_y(xp[0]) - mnbF) * idb;
            float c2 = xp[2], c3 = xp[3], c4 = xp[4], c5 = xp[5];
            if (s == 1 && l == 0) {
                if (spd != 0.f) c2 = spd;
                if (vx  != 0.f) c4 = vx;
                if (vy  != 0.f) c5 = vy;
            }
            op[2] = c2; op[3] = c3; op[4] = c4; op[5] = c5;
        }
    }
    if (tid < 2) {
        const float* xp = (tid == 0) ? (x1 + ((size_t)b * LL + (LL - 1)) * CH)
                                     : (x2 + ((size_t)b * LL + 0) * CH);
        float* op = xn3 + ((size_t)b * 2 + tid) * CH;
        float lat = xp[0], lon = xp[1];
        float c2 = xp[2], c3 = xp[3], c4 = xp[4], c5 = xp[5];
        if (tid == 1) {
            if (spd != 0.f) c2 = spd;
            if (vx  != 0.f) c4 = vx;
            if (vy  != 0.f) c5 = vy;
        }
        op[0] = (merc_y(lat) - mnaF) * ida;   // reference quirk preserved
        op[1] = (merc_x(lon) - mnbF) * idb;
        op[2] = c2; op[3] = c3; op[4] = c4; op[5] = c5;
    }
}

// ---------------------------------------------------------------------------
// Kernel: conv features -> bf16[*,224]
// ---------------------------------------------------------------------------
__global__ __launch_bounds__(256)
void k_conv_feat(const float* __restrict__ xn, int Lseq, uint16_t* __restrict__ feat,
                 const float* __restrict__ w1, const float* __restrict__ b1,
                 const float* __restrict__ w3, const float* __restrict__ b3,
                 const float* __restrict__ w5, const float* __restrict__ b5) {
    __shared__ float sw1[64 * 6], sw3[64 * 6 * 3], sw5[64 * 6 * 5];
    __shared__ float sb1[64], sb3[64], sb5[64];
    int tid = threadIdx.x;
    for (int i = tid; i < 64 * 6;     i += 256) sw1[i] = w1[i];
    for (int i = tid; i < 64 * 6 * 3; i += 256) sw3[i] = w3[i];
    for (int i = tid; i < 64 * 6 * 5; i += 256) sw5[i] = w5[i];
    for (int i = tid; i < 64; i += 256) { sb1[i] = b1[i]; sb3[i] = b3[i]; sb5[i] = b5[i]; }
    __syncthreads();

    int idx = blockIdx.x * 256 + tid;
    if (idx >= BB * Lseq) return;
    int b = idx / Lseq, l = idx % Lseq;

    float xv[5][6];
    for (int dp = -2; dp <= 2; ++dp) {
        int ll = l + dp;
        for (int c = 0; c < 6; ++c)
            xv[dp + 2][c] = (ll >= 0 && ll < Lseq) ? xn[((size_t)b * Lseq + ll) * CH + c] : 0.f;
    }
    uint16_t* fp = feat + (size_t)idx * FD;
    for (int c = 0; c < 6; ++c) fp[c] = f2bf(xv[2][c]);
    for (int o = 0; o < 64; ++o) {
        float s1 = sb1[o], s3 = sb3[o], s5 = sb5[o];
        for (int c = 0; c < 6; ++c) {
            s1 += sw1[o * 6 + c] * xv[2][c];
            for (int k = 0; k < 3; ++k) s3 += sw3[(o * 6 + c) * 3 + k] * xv[1 + k][c];
            for (int k = 0; k < 5; ++k) s5 += sw5[(o * 6 + c) * 5 + k] * xv[k][c];
        }
        fp[6 + o]   = f2bf(fmaxf(s1, 0.f));
        fp[70 + o]  = f2bf(fmaxf(s3, 0.f));
        fp[134 + o] = f2bf(fmaxf(s5, 0.f));
    }
    for (int k = 198; k < FD; ++k) fp[k] = 0;
}

// ---------------------------------------------------------------------------
// Kernel: pack f32 weight [G=1024, Kreal] into WMMA B-fragment tiles.
// Layout: tile (nt, kt) at offset (nt*KT + kt)*512 (nt-major) so each WG's
// w_hh slice {nt, nt+16, nt+32, nt+48} is 4 contiguous 8KB rows -> 2-D TDM.
// ---------------------------------------------------------------------------
__global__ void k_pack_w(const float* __restrict__ w, int Kreal, int KT,
                         uint16_t* __restrict__ Bp) {
    int e = blockIdx.x * 256 + threadIdx.x;
    int total = KT * 64 * 512;
    if (e >= total) return;
    int tile = e / 512;
    int within = e % 512;
    int nt = tile / KT;
    int kt = tile % KT;
    int lane = within / 16;
    int j = within % 16;
    int K = kt * 32 + ((lane < 16) ? 0 : 8) + ((j < 8) ? j : j + 8);
    int N = nt * 16 + (lane & 15);
    float v = (K < Kreal) ? w[(size_t)N * Kreal + K] : 0.f;
    Bp[e] = f2bf(v);
}

// ---------------------------------------------------------------------------
// Kernel: GEMM  C[M,1024] = A[M,K] @ Bsw + (bias1+bias2), bf16 in/out, f32 acc.
// Software-pipelined: next k-tile's fragments prefetched during current WMMAs.
// ---------------------------------------------------------------------------
__global__ __launch_bounds__(256)
void k_gemm_xg(const uint16_t* __restrict__ A, unsigned long lda,
               const uint16_t* __restrict__ Bp, int KT,
               const float* __restrict__ bias1, const float* __restrict__ bias2,
               uint16_t* __restrict__ C, unsigned long ldc) {
    int wave = threadIdx.x >> 5;
    int lane = threadIdx.x & 31;
    int rowbase = blockIdx.x * 128 + wave * 16;
    int ntbase = blockIdx.y * 4;

    v8f acc0 = {0,0,0,0,0,0,0,0};
    v8f acc1 = acc0, acc2 = acc0, acc3 = acc0;

    v16bf aC  = load_a_frag(A, lda, rowbase, 0);
    v16bf bC0 = load_b_frag(Bp + ((size_t)((ntbase + 0) * KT + 0)) * 512);
    v16bf bC1 = load_b_frag(Bp + ((size_t)((ntbase + 1) * KT + 0)) * 512);
    v16bf bC2 = load_b_frag(Bp + ((size_t)((ntbase + 2) * KT + 0)) * 512);
    v16bf bC3 = load_b_frag(Bp + ((size_t)((ntbase + 3) * KT + 0)) * 512);

    for (int kt = 0; kt < KT; ++kt) {
        v16bf aN = aC, bN0 = bC0, bN1 = bC1, bN2 = bC2, bN3 = bC3;
        int ktn = kt + 1;
        if (ktn < KT) {
            aN  = load_a_frag(A, lda, rowbase, ktn);
            bN0 = load_b_frag(Bp + ((size_t)((ntbase + 0) * KT + ktn)) * 512);
            bN1 = load_b_frag(Bp + ((size_t)((ntbase + 1) * KT + ktn)) * 512);
            bN2 = load_b_frag(Bp + ((size_t)((ntbase + 2) * KT + ktn)) * 512);
            bN3 = load_b_frag(Bp + ((size_t)((ntbase + 3) * KT + ktn)) * 512);
        }
        acc0 = __builtin_amdgcn_wmma_f32_16x16x32_bf16(false, aC, false, bC0, (short)0, acc0, false, false);
        acc1 = __builtin_amdgcn_wmma_f32_16x16x32_bf16(false, aC, false, bC1, (short)0, acc1, false, false);
        acc2 = __builtin_amdgcn_wmma_f32_16x16x32_bf16(false, aC, false, bC2, (short)0, acc2, false, false);
        acc3 = __builtin_amdgcn_wmma_f32_16x16x32_bf16(false, aC, false, bC3, (short)0, acc3, false, false);
        aC = aN; bC0 = bN0; bC1 = bN1; bC2 = bN2; bC3 = bN3;
    }
    int n15 = lane & 15;
    int mo = (lane < 16) ? 0 : 8;
    v8f* accs[4] = {&acc0, &acc1, &acc2, &acc3};
    for (int t = 0; t < 4; ++t) {
        int n = (ntbase + t) * 16 + n15;
        float bb = bias1[n] + bias2[n];
        v8f av = *accs[t];
#pragma unroll
        for (int r = 0; r < 8; ++r)
            C[(size_t)(rowbase + mo + r) * ldc + n] = f2bf(av[r] + bb);
    }
}

// ---------------------------------------------------------------------------
// Gang-of-16-WGs spin barrier in global memory
// ---------------------------------------------------------------------------
__device__ __forceinline__ void grid_sync(int* bar, int nblk) {
    __syncthreads();
    __threadfence();
    if (threadIdx.x == 0) {
        volatile int* vb = (volatile int*)bar;
        int gen = vb[1];
        if (atomicAdd(&bar[0], 1) == nblk - 1) {
            bar[0] = 0;
            __threadfence();
            atomicAdd(&bar[1], 1);
        } else {
            while (vb[1] == gen) { __builtin_amdgcn_s_sleep(2); }
        }
    }
    __syncthreads();
    __threadfence();
}

// ---------------------------------------------------------------------------
// Kernel: forward LSTM recurrence. 16 WGs per sequence; WG w owns hidden
// columns [16w,16w+16) and gate tiles {w,w+16,w+32,w+48}.
// w_hh slice: one 2-D TDM into static LDS at start.
// h matrix:   one 1-D TDM (64 KB) into dynamic LDS every step; A-frags via ds.
// Cell state lives in registers in the WMMA C-fragment layout.
// ---------------------------------------------------------------------------
__global__ __launch_bounds__(256)
void k_lstm_fwd(const uint16_t* __restrict__ xg, int L, int Lstride,
                const uint16_t* __restrict__ WhhP,
                uint16_t* __restrict__ hA, uint16_t* __restrict__ hB,
                float* __restrict__ hfin, int* bar) {
    __shared__ __align__(16) uint16_t sW[4 * KT_HH * 512];   // 32 KB
    extern __shared__ uint16_t sH[];                         // 64 KB dynamic: h matrix
    int w = blockIdx.x;

    // stage this WG's w_hh slice: 4 rows (quadrants) x 8 KB, row stride 128 KB
#if HAVE_TDM
    if (threadIdx.x < 32) {
        tdm_load_2d(WhhP + (size_t)w * KT_HH * 512, sW,
                    (KT_HH * 512 * 2) / 8, 4, (16 * KT_HH * 512 * 2) / 8);
        __builtin_amdgcn_s_wait_tensorcnt(0);
    }
    __syncthreads();
#else
    for (int i = threadIdx.x; i < 4 * KT_HH * 512; i += 256) {
        int q = i >> 12;
        int rem = i & 4095;
        int kt = rem >> 9;
        int e = rem & 511;
        int nt = w + q * 16;
        sW[i] = WhhP[((size_t)(nt * KT_HH + kt)) * 512 + e];
    }
    __syncthreads();
#endif

    int wave = threadIdx.x >> 5;
    int lane = threadIdx.x & 31;
    int rowbase = wave * 16;
    int n15 = lane & 15;
    int mo = (lane < 16) ? 0 : 8;
    int hcol = w * 16 + n15;

    v8f cst = {0,0,0,0,0,0,0,0};

    for (int t = 0; t < L; ++t) {
        const uint16_t* hr = (t & 1) ? hB : hA;
        uint16_t* hw = (t & 1) ? hA : hB;

        // pull full h (128x256 bf16, contiguous 64 KB) into LDS via TDM
#if HAVE_TDM
        if (threadIdx.x < 32) {
            tdm_load_2d(hr, sH, (BB * HH * 2) / 8, 1, 0);
            __builtin_amdgcn_s_wait_tensorcnt(0);
        }
        __syncthreads();
#else
        for (int i = threadIdx.x; i < (BB * HH) / 8; i += 256)
            ((uint4*)sH)[i] = ((const uint4*)hr)[i];
        __syncthreads();
#endif

        v8f gi = {0,0,0,0,0,0,0,0};
        v8f gf = gi, gg = gi, go = gi;
#pragma unroll
        for (int kt = 0; kt < KT_HH; ++kt) {
            v16bf a  = load_a_frag(sH, HH, rowbase, kt);
            v16bf bi = load_b_frag(sW + (0 * KT_HH + kt) * 512);
            v16bf bf = load_b_frag(sW + (1 * KT_HH + kt) * 512);
            v16bf bg = load_b_frag(sW + (2 * KT_HH + kt) * 512);
            v16bf bo = load_b_frag(sW + (3 * KT_HH + kt) * 512);
            gi = __builtin_amdgcn_wmma_f32_16x16x32_bf16(false, a, false, bi, (short)0, gi, false, false);
            gf = __builtin_amdgcn_wmma_f32_16x16x32_bf16(false, a, false, bf, (short)0, gf, false, false);
            gg = __builtin_amdgcn_wmma_f32_16x16x32_bf16(false, a, false, bg, (short)0, gg, false, false);
            go = __builtin_amdgcn_wmma_f32_16x16x32_bf16(false, a, false, bo, (short)0, go, false, false);
        }
#pragma unroll
        for (int r = 0; r < 8; ++r) {
            int row = rowbase + mo + r;                         // batch index
            const uint16_t* xp = xg + ((size_t)row * Lstride + t) * (size_t)GG;
            float iv = gi[r] + bf2f(xp[hcol]);
            float fv = gf[r] + bf2f(xp[256 + hcol]);
            float gv = gg[r] + bf2f(xp[512 + hcol]);
            float ov = go[r] + bf2f(xp[768 + hcol]);
            float cc = sigf(fv) * cst[r] + sigf(iv) * tanhf(gv);
            cst[r] = cc;
            float hh = sigf(ov) * tanhf(cc);
            hw[(size_t)row * HH + hcol] = f2bf(hh);
            if (t == L - 1) hfin[(size_t)row * HH + hcol] = hh;
        }
        grid_sync(bar, 16);
    }
}

// ---------------------------------------------------------------------------
// Kernel: backward direction = single LSTM step from zero state (biases folded)
// ---------------------------------------------------------------------------
__global__ void k_tail_bwd(const uint16_t* __restrict__ xgb, float* __restrict__ hfinb) {
    int i = blockIdx.x * 256 + threadIdx.x;
    int b = i >> 8, j = i & 255;
    const uint16_t* xp = xgb + (size_t)b * GG;
    float iv = bf2f(xp[j]);
    float gv = bf2f(xp[512 + j]);
    float ov = bf2f(xp[768 + j]);
    float cc = sigf(iv) * tanhf(gv);
    hfinb[i] = sigf(ov) * tanhf(cc);
}

// ---------------------------------------------------------------------------
// Kernel: head
// ---------------------------------------------------------------------------
__global__ __launch_bounds__(256)
void k_head(const float* __restrict__ h1f, const float* __restrict__ h1b,
            const float* __restrict__ h2f, const float* __restrict__ h2b,
            const float* __restrict__ h3f, const float* __restrict__ h3b,
            const float* __restrict__ fc1w, const float* __restrict__ fc1b,
            const float* __restrict__ fc2w, const float* __restrict__ fc2b,
            float* __restrict__ out) {
    __shared__ float d[1024];
    __shared__ float s1[512];
    __shared__ float red[256];
    int b = blockIdx.x, tid = threadIdx.x;
    for (int k = tid; k < 256; k += 256) {
        size_t o = (size_t)b * 256 + k;
        d[k]       = fabsf(h1f[o] - h2f[o]);
        d[256 + k] = fabsf(h1b[o] - h2b[o]);
        d[512 + k] = fabsf(h1f[o] - h3f[o]);
        d[768 + k] = fabsf(h1b[o] - h3b[o]);
    }
    __syncthreads();
    for (int o = tid; o < 512; o += 256) {
        float s = fc1b[o];
        const float* wr = fc1w + (size_t)o * 1024;
        for (int k = 0; k < 1024; ++k) s += wr[k] * d[k];
        s1[o] = fmaxf(s, 0.f);
    }
    __syncthreads();
    float p = 0.f;
    for (int o = tid; o < 512; o += 256) p += s1[o] * fc2w[o];
    red[tid] = p;
    __syncthreads();
    for (int st = 128; st > 0; st >>= 1) {
        if (tid < st) red[tid] += red[tid + st];
        __syncthreads();
    }
    if (tid == 0) out[b] = 1.0f / (1.0f + expf(-(red[0] + fc2b[0])));
}

// ---------------------------------------------------------------------------
// Host launcher
// ---------------------------------------------------------------------------
extern "C" void kernel_launch(void* const* d_in, const int* in_sizes, int n_in,
                              void* d_out, int out_size, void* d_ws, size_t ws_size,
                              hipStream_t stream) {
    (void)in_sizes; (void)n_in; (void)out_size; (void)ws_size;
    const float* x1     = (const float*)d_in[0];
    const float* x2     = (const float*)d_in[1];
    const float* dtime  = (const float*)d_in[2];
    const float* conv1w = (const float*)d_in[3];
    const float* conv1b = (const float*)d_in[4];
    const float* conv3w = (const float*)d_in[5];
    const float* conv3b = (const float*)d_in[6];
    const float* conv5w = (const float*)d_in[7];
    const float* conv5b = (const float*)d_in[8];
    const float* wihf   = (const float*)d_in[9];
    const float* whhf   = (const float*)d_in[10];
    const float* bihf   = (const float*)d_in[11];
    const float* bhhf   = (const float*)d_in[12];
    const float* wihb   = (const float*)d_in[13];
    const float* whhb   = (const float*)d_in[14];
    const float* bihb   = (const float*)d_in[15];
    const float* bhhb   = (const float*)d_in[16];
    const float* fc1w   = (const float*)d_in[17];
    const float* fc1b   = (const float*)d_in[18];
    const float* fc2w   = (const float*)d_in[19];
    const float* fc2b   = (const float*)d_in[20];
    float* out = (float*)d_out;

    uint8_t* ws = (uint8_t*)d_ws;
    size_t off = 0;
    auto alloc = [&](size_t bytes) -> void* {
        void* p = (void*)(ws + off);
        off += (bytes + 255) & ~(size_t)255;
        return p;
    };

    float* xn1 = (float*)alloc((size_t)BB * LL * CH * 4);
    float* xn2 = (float*)alloc((size_t)BB * LL * CH * 4);
    float* xn3 = (float*)alloc((size_t)BB * 2 * CH * 4);

    uint16_t* feat1 = (uint16_t*)alloc((size_t)BB * LL * FD * 2);
    uint16_t* feat2 = (uint16_t*)alloc((size_t)BB * LL * FD * 2);
    uint16_t* feat3 = (uint16_t*)alloc((size_t)BB * 2 * FD * 2);

    uint16_t* pWihF = (uint16_t*)alloc((size_t)KT_IH * 64 * 512 * 2);
    uint16_t* pWihB = (uint16_t*)alloc((size_t)KT_IH * 64 * 512 * 2);
    uint16_t* pWhhF = (uint16_t*)alloc((size_t)KT_HH * 64 * 512 * 2);
    uint16_t* pWhhB = (uint16_t*)alloc((size_t)KT_HH * 64 * 512 * 2);

    uint16_t* xgf1 = (uint16_t*)alloc((size_t)BB * LL * GG * 2);   // 128 MB
    uint16_t* xgf2 = (uint16_t*)alloc((size_t)BB * LL * GG * 2);   // 128 MB
    uint16_t* xgf3 = (uint16_t*)alloc((size_t)BB * 2 * GG * 2);
    uint16_t* xgb1 = (uint16_t*)alloc((size_t)BB * GG * 2);
    uint16_t* xgb2 = (uint16_t*)alloc((size_t)BB * GG * 2);
    uint16_t* xgb3 = (uint16_t*)alloc((size_t)BB * GG * 2);

    uint16_t* hbuf = (uint16_t*)alloc((size_t)3 * 2 * BB * HH * 2); // ping-pong h
    int*      bars = (int*)alloc(256);

    float* hfF1 = (float*)alloc((size_t)BB * HH * 4);
    float* hfF2 = (float*)alloc((size_t)BB * HH * 4);
    float* hfF3 = (float*)alloc((size_t)BB * HH * 4);
    float* hfB1 = (float*)alloc((size_t)BB * HH * 4);
    float* hfB2 = (float*)alloc((size_t)BB * HH * 4);
    float* hfB3 = (float*)alloc((size_t)BB * HH * 4);

    // --- re-init mutable state every call ---
    k_zero_u32<<<64, 256, 0, stream>>>((unsigned*)hbuf, (size_t)3 * 2 * BB * HH / 2);
    k_zero_u32<<<1, 64, 0, stream>>>((unsigned*)bars, 64);

    // --- feature prep ---
    k_prep<<<BB, 256, 0, stream>>>(x1, x2, dtime, xn1, xn2, xn3);
    k_conv_feat<<<(BB * LL) / 256, 256, 0, stream>>>(xn1, LL, feat1,
        conv1w, conv1b, conv3w, conv3b, conv5w, conv5b);
    k_conv_feat<<<(BB * LL) / 256, 256, 0, stream>>>(xn2, LL, feat2,
        conv1w, conv1b, conv3w, conv3b, conv5w, conv5b);
    k_conv_feat<<<1, 256, 0, stream>>>(xn3, 2, feat3,
        conv1w, conv1b, conv3w, conv3b, conv5w, conv5b);

    // --- weight packing ---
    k_pack_w<<<(KT_IH * 64 * 512 + 255) / 256, 256, 0, stream>>>(wihf, 198, KT_IH, pWihF);
    k_pack_w<<<(KT_IH * 64 * 512 + 255) / 256, 256, 0, stream>>>(wihb, 198, KT_IH, pWihB);
    k_pack_w<<<(KT_HH * 64 * 512 + 255) / 256, 256, 0, stream>>>(whhf, 256, KT_HH, pWhhF);
    k_pack_w<<<(KT_HH * 64 * 512 + 255) / 256, 256, 0, stream>>>(whhb, 256, KT_HH, pWhhB);

    // --- input projections (WMMA GEMMs), biases folded in ---
    k_gemm_xg<<<dim3(512, 16), 256, 0, stream>>>(feat1, (unsigned long)FD, pWihF, KT_IH,
                                                 bihf, bhhf, xgf1, (unsigned long)GG);
    k_gemm_xg<<<dim3(512, 16), 256, 0, stream>>>(feat2, (unsigned long)FD, pWihF, KT_IH,
                                                 bihf, bhhf, xgf2, (unsigned long)GG);
    k_gemm_xg<<<dim3(2, 16), 256, 0, stream>>>(feat3, (unsigned long)FD, pWihF, KT_IH,
                                               bihf, bhhf, xgf3, (unsigned long)GG);
    k_gemm_xg<<<dim3(1, 16), 256, 0, stream>>>(feat1 + (size_t)(LL - 1) * FD,
                                               (unsigned long)LL * FD, pWihB, KT_IH,
                                               bihb, bhhb, xgb1, (unsigned long)GG);
    k_gemm_xg<<<dim3(1, 16), 256, 0, stream>>>(feat2 + (size_t)(LL - 1) * FD,
                                               (unsigned long)LL * FD, pWihB, KT_IH,
                                               bihb, bhhb, xgb2, (unsigned long)GG);
    k_gemm_xg<<<dim3(1, 16), 256, 0, stream>>>(feat3 + (size_t)1 * FD,
                                               (unsigned long)2 * FD, pWihB, KT_IH,
                                               bihb, bhhb, xgb3, (unsigned long)GG);

    // --- forward recurrences (3 concurrent gangs of 16 WGs; 64 KB dynamic LDS) ---
    uint16_t* hA1 = hbuf + 0 * 2 * BB * HH;
    uint16_t* hA2 = hbuf + 1 * 2 * BB * HH;
    uint16_t* hA3 = hbuf + 2 * 2 * BB * HH;
    size_t dynLds = (size_t)BB * HH * 2;  // 64 KB
    k_lstm_fwd<<<16, 256, dynLds, stream>>>(xgf1, LL, LL, pWhhF, hA1, hA1 + BB * HH, hfF1, bars + 0);
    k_lstm_fwd<<<16, 256, dynLds, stream>>>(xgf2, LL, LL, pWhhF, hA2, hA2 + BB * HH, hfF2, bars + 8);
    k_lstm_fwd<<<16, 256, dynLds, stream>>>(xgf3, 2, 2, pWhhF, hA3, hA3 + BB * HH, hfF3, bars + 16);

    // --- backward single steps ---
    k_tail_bwd<<<(BB * HH) / 256, 256, 0, stream>>>(xgb1, hfB1);
    k_tail_bwd<<<(BB * HH) / 256, 256, 0, stream>>>(xgb2, hfB2);
    k_tail_bwd<<<(BB * HH) / 256, 256, 0, stream>>>(xgb3, hfB3);

    // --- head ---
    k_head<<<BB, 256, 0, stream>>>(hfF1, hfB1, hfF2, hfB2, hfF3, hfB3,
                                   fc1w, fc1b, fc2w, fc2b, out);
}
